// GRU_90426241450185
// MI455X (gfx1250) — matrix operand order repaired and checked by reading, not compiled, not run on hardware
//
#include <hip/hip_runtime.h>

// ---------------------------------------------------------------------------
// Types for gfx1250 WMMA (wave32)
// ---------------------------------------------------------------------------
typedef __attribute__((ext_vector_type(16))) __bf16      v16bf;
typedef __attribute__((ext_vector_type(8)))  float       v8f;
typedef __attribute__((ext_vector_type(4)))  unsigned int u32x4;

union Frag16 {
    v16bf v;
    u32x4 u[2];   // 32 bytes = 16 bf16
};

#define BB   8
#define SS   2048
#define DIN  1024
#define DST  1024
#define HH   16
#define DH   64
#define DOUT 1024
#define MROWS (BB * SS)          // 16384
#define PROJW (4 * DST)          // 4096

__device__ __forceinline__ unsigned short f32_to_bf16(float f) {
    unsigned int u = __float_as_uint(f);
    u += 0x7FFFu + ((u >> 16) & 1u);          // round to nearest even
    return (unsigned short)(u >> 16);
}

__device__ __forceinline__ float sigmoid_fast(float x) {
    return __frcp_rn(1.0f + __expf(-x));
}
__device__ __forceinline__ float tanh_fast(float x) {
    float e = __expf(-2.0f * fabsf(x));
    float t = (1.0f - e) * __frcp_rn(1.0f + e);
    return copysignf(t, x);
}

// ---------------------------------------------------------------------------
// Elementwise fp32 -> bf16 (row-major, used for A matrices)
// ---------------------------------------------------------------------------
__global__ void cvt_bf16_kernel(const float* __restrict__ src,
                                unsigned short* __restrict__ dst, size_t n) {
    size_t i = (size_t)blockIdx.x * blockDim.x + threadIdx.x;
    if (i < n) dst[i] = f32_to_bf16(src[i]);
}

// ---------------------------------------------------------------------------
// Pack row-major fp32 B[K,N] into WMMA-B-friendly bf16 layout Bp[K/32][N][32]
// (each B fragment becomes two contiguous b128 loads per lane)
// ---------------------------------------------------------------------------
__global__ void pack_b_kernel(const float* __restrict__ B,
                              unsigned short* __restrict__ Bp, int K, int N) {
    size_t idx = (size_t)blockIdx.x * blockDim.x + threadIdx.x;
    if (idx >= (size_t)K * N) return;
    int ki = (int)(idx & 31);
    size_t t = idx >> 5;
    int n  = (int)(t % N);
    int kc = (int)(t / N);
    Bp[idx] = f32_to_bf16(B[(size_t)(kc * 32 + ki) * N + n]);
}

// ---------------------------------------------------------------------------
// Transpose state weights: Wt[m][e][d] = bf16(sw[m][d][e]); m in [0,3H)
// Row e contiguous in d -> B-fragment loads are contiguous b128s.
// ---------------------------------------------------------------------------
__global__ void pack_w_kernel(const float* __restrict__ sw,
                              unsigned short* __restrict__ Wt) {
    int idx = blockIdx.x * blockDim.x + threadIdx.x;   // 3H*64*64 = 196608
    if (idx >= 3 * HH * DH * DH) return;
    int d = idx & 63;
    int e = (idx >> 6) & 63;
    int m = idx >> 12;
    Wt[idx] = f32_to_bf16(sw[((size_t)m * DH + d) * DH + e]);
}

// ---------------------------------------------------------------------------
// bf16 WMMA GEMM: C[M,N] = A[M,K] @ B[K,N]
//   A: row-major bf16; Bp: packed [K/32][N][32] bf16; C: row-major f32
//   Block = 128 threads (4 waves, 2x2); block tile = 128(M) x 128(N);
//   wave tile = 64(M) x 64(N) = 4 x 4 WMMA tiles, K chunked by 32.
//   16 WMMAs per chunk give the scheduler enough independent matrix ops to
//   cover WMMA->VALU WAR hazard windows (fewer v_nop bubbles), and each
//   fragment load now feeds 4 WMMAs (1.0 load/WMMA vs 1.5 before).
// ---------------------------------------------------------------------------
__global__ __launch_bounds__(128)
void gemm_bf16_kernel(const unsigned short* __restrict__ A,
                      const unsigned short* __restrict__ Bp,
                      float* __restrict__ C, int M, int N, int K) {
    const int lane = threadIdx.x & 31;
    const int wave = threadIdx.x >> 5;
    const int row  = lane & 15;
    const int hi   = lane >> 4;                 // 0: lanes 0-15, 1: lanes 16-31
    const int m0 = blockIdx.y * 128 + (wave >> 1) * 64;
    const int n0 = blockIdx.x * 128 + (wave & 1) * 64;

    v8f acc[4][4] = {};

    for (int kc = 0; kc < K; kc += 32) {
        Frag16 a[4];
#pragma unroll
        for (int f = 0; f < 4; f++) {
            const unsigned short* ap =
                A + (size_t)(m0 + f * 16 + row) * K + kc + hi * 16;
            a[f].u[0] = *(const u32x4*)(ap);
            a[f].u[1] = *(const u32x4*)(ap + 8);
        }
#pragma unroll
        for (int nt = 0; nt < 4; nt++) {
            const unsigned short* bp =
                Bp + ((size_t)(kc >> 5) * N + (n0 + nt * 16 + row)) * 32 + hi * 16;
            Frag16 b;
            b.u[0] = *(const u32x4*)(bp);
            b.u[1] = *(const u32x4*)(bp + 8);
#pragma unroll
            for (int mi = 0; mi < 4; mi++)
                acc[mi][nt] = __builtin_amdgcn_wmma_f32_16x16x32_bf16(
                    false, a[mi].v, false, b.v, (short)0, acc[mi][nt],
                    false, false);
        }
    }

#pragma unroll
    for (int f = 0; f < 4; f++)
#pragma unroll
        for (int nt = 0; nt < 4; nt++)
#pragma unroll
            for (int i = 0; i < 8; i++) {
                int m = m0 + f * 16 + hi * 8 + i;
                int n = n0 + nt * 16 + row;
                C[(size_t)m * N + n] = acc[f][nt][i];
            }
}

// ---------------------------------------------------------------------------
// Recurrent scan: one block (128 thr = 4 waves) per head.
//   Wave w owns N-tile nt = w (16 output dims); M = batch (8 valid rows).
//   Per-wave recurrence weights (3 gates x 2 K-frags = 48 VGPRs) stay in
//   registers for the whole scan. LDS holds only the 16x64 bf16 h / r*h
//   shuffle buffers used to re-shape D-layout results into A fragments.
//   x-terms (gate pre-activations from proj) are register double-buffered.
// ---------------------------------------------------------------------------
__device__ __forceinline__ v8f load_xterm(const float* __restrict__ proj,
                                          int s, int head, int gate_off,
                                          int nt, int row, int hi) {
    v8f acc;
#pragma unroll
    for (int i = 0; i < 8; i++) {
        int b = (hi * 8 + i) & 7;   // lanes 16-31 are padding rows: reuse valid b
        acc[i] = proj[((size_t)b * SS + s) * PROJW +
                      gate_off + head * DH + nt * 16 + row];
    }
    return acc;
}

__device__ __forceinline__ v8f gate2(const Frag16 (&a)[2], const Frag16 (&b)[2],
                                     v8f acc) {
    acc = __builtin_amdgcn_wmma_f32_16x16x32_bf16(
        false, a[0].v, false, b[0].v, (short)0, acc, false, false);
    acc = __builtin_amdgcn_wmma_f32_16x16x32_bf16(
        false, a[1].v, false, b[1].v, (short)0, acc, false, false);
    return acc;
}

__device__ __forceinline__ void load_afrag(Frag16 (&a)[2],
                                           const unsigned short* buf,
                                           int row, int hi) {
#pragma unroll
    for (int kc = 0; kc < 2; kc++) {
        const unsigned short* p = buf + row * 64 + kc * 32 + hi * 16;
        a[kc].u[0] = *(const u32x4*)(p);
        a[kc].u[1] = *(const u32x4*)(p + 8);
    }
}

__global__ __launch_bounds__(128)
void scan_kernel(const float* __restrict__ proj,
                 const unsigned short* __restrict__ Wt,
                 float* __restrict__ h_seq) {
    __shared__ unsigned short hbuf[16 * 64];
    __shared__ unsigned short rbuf[16 * 64];

    const int tid  = threadIdx.x;
    const int lane = tid & 31;
    const int nt   = tid >> 5;           // wave id == owned N-tile
    const int head = blockIdx.x;
    const int row  = lane & 15;
    const int hi   = lane >> 4;

    // Persistent register-resident B fragments: gate g (0=W cand, 1=Wf, 2=Wr)
    Frag16 bw[3][2];
#pragma unroll
    for (int g = 0; g < 3; g++)
#pragma unroll
        for (int kc = 0; kc < 2; kc++) {
            const unsigned short* wp = Wt + (size_t)(g * HH + head) * 4096 +
                                       (nt * 16 + row) * 64 + kc * 32 + hi * 16;
            bw[g][kc].u[0] = *(const u32x4*)(wp);
            bw[g][kc].u[1] = *(const u32x4*)(wp + 8);
        }

    for (int i = tid; i < 1024; i += 128) hbuf[i] = 0;
    __syncthreads();

    v8f h_d = {};                        // D-layout h (f32) for this N-tile
    v8f xi_c = load_xterm(proj, 0, head, 0 * DST, nt, row, hi);
    v8f xf_c = load_xterm(proj, 0, head, 1 * DST, nt, row, hi);
    v8f xr_c = load_xterm(proj, 0, head, 2 * DST, nt, row, hi);

#pragma unroll 1
    for (int s = 0; s < SS; s++) {
        // ---- A fragments of h (full hbuf; published before last barrier)
        Frag16 ah[2];
        load_afrag(ah, hbuf, row, hi);

        // ---- prefetch next step's gate pre-activations (off critical path)
        const int sn = (s + 1 < SS) ? s + 1 : s;
        v8f xi_n = load_xterm(proj, sn, head, 0 * DST, nt, row, hi);
        v8f xf_n = load_xterm(proj, sn, head, 1 * DST, nt, row, hi);
        v8f xr_n = load_xterm(proj, sn, head, 2 * DST, nt, row, hi);

        // ---- r = sigmoid(xr + h @ Wr), f = sigmoid(xf + h @ Wf)
        v8f ar = gate2(ah, bw[2], xr_c);
        v8f af = gate2(ah, bw[1], xf_c);
        v8f rg, fg;
#pragma unroll
        for (int i = 0; i < 8; i++) {
            rg[i] = sigmoid_fast(ar[i]);
            fg[i] = sigmoid_fast(af[i]);
        }

        // ---- publish r * h slice (bf16, [m][d]) for the c-gate A operand
#pragma unroll
        for (int i = 0; i < 8; i++) {
            int m = hi * 8 + i;
            rbuf[m * 64 + nt * 16 + row] = f32_to_bf16(rg[i] * h_d[i]);
        }
        __syncthreads();

        // ---- c = tanh(xi + (r*h) @ W);  h_new = f*h + (1-f)*c
        Frag16 arh[2];
        load_afrag(arh, rbuf, row, hi);
        v8f ac = gate2(arh, bw[0], xi_c);
#pragma unroll
        for (int i = 0; i < 8; i++) {
            float c  = tanh_fast(ac[i]);
            float ff = fg[i];
            float hn = ff * h_d[i] + (1.0f - ff) * c;
            h_d[i] = hn;
            int m = hi * 8 + i;
            hbuf[m * 64 + nt * 16 + row] = f32_to_bf16(hn);
        }

        // ---- emit h_seq (valid batches live in lanes 0-15, m = i)
        if (hi == 0) {
#pragma unroll
            for (int i = 0; i < 8; i++)
                h_seq[((size_t)i * SS + s) * DST + head * DH + nt * 16 + row] =
                    h_d[i];
        }

        xi_c = xi_n; xf_c = xf_n; xr_c = xr_n;
        __syncthreads();                 // hbuf ready for next iteration
    }
}

// ---------------------------------------------------------------------------
// y = rmsnorm(h * silu(g)) * norm_w  ->  bf16 rows for the output GEMM
// One block (256 thr) per (b,s) row of 1024 elements.
// ---------------------------------------------------------------------------
__global__ __launch_bounds__(256)
void gate_norm_kernel(const float* __restrict__ h_seq,
                      const float* __restrict__ proj,
                      const float* __restrict__ nw,
                      unsigned short* __restrict__ y) {
    __shared__ float red[8];
    const size_t rowi = blockIdx.x;               // b*S + s
    const float* hr = h_seq + rowi * DST;
    const float* gr = proj + rowi * PROJW + 3 * DST;

    float v[4];
    float ss = 0.0f;
#pragma unroll
    for (int j = 0; j < 4; j++) {
        int d = threadIdx.x + j * 256;
        float g  = gr[d];
        float t  = hr[d] * g * sigmoid_fast(g);   // h * silu(g)
        v[j] = t;
        ss += t * t;
    }
#pragma unroll
    for (int o = 16; o > 0; o >>= 1) ss += __shfl_xor(ss, o, 32);
    if ((threadIdx.x & 31) == 0) red[threadIdx.x >> 5] = ss;
    __syncthreads();
    float tot = 0.0f;
#pragma unroll
    for (int w = 0; w < 8; w++) tot += red[w];
    float scale = __frsqrt_rn(tot * (1.0f / (float)DST) + 1e-6f);
#pragma unroll
    for (int j = 0; j < 4; j++) {
        int d = threadIdx.x + j * 256;
        y[rowi * DST + d] = f32_to_bf16(v[j] * scale * nw[d]);
    }
}

// ---------------------------------------------------------------------------
// Launch
// ---------------------------------------------------------------------------
extern "C" void kernel_launch(void* const* d_in, const int* in_sizes, int n_in,
                              void* d_out, int out_size, void* d_ws, size_t ws_size,
                              hipStream_t stream) {
    const float* x    = (const float*)d_in[0];   // [B,S,1024]
    const float* w_in = (const float*)d_in[1];   // [1024,4096]
    const float* sw   = (const float*)d_in[2];   // [48,64,64]
    const float* nw   = (const float*)d_in[3];   // [1024]
    const float* w_o  = (const float*)d_in[4];   // [1024,1024]
    float* out = (float*)d_out;                  // [B,S,1024] f32

    char* ws = (char*)d_ws;
    size_t off = 0;
    auto alloc = [&](size_t bytes) {
        void* p = ws + off;
        off += (bytes + 255) & ~(size_t)255;
        return p;
    };
    unsigned short* xb   = (unsigned short*)alloc((size_t)MROWS * DIN * 2);
    unsigned short* wip  = (unsigned short*)alloc((size_t)DIN * PROJW * 2);
    float*          proj = (float*)        alloc((size_t)MROWS * PROJW * 4);
    unsigned short* wt   = (unsigned short*)alloc((size_t)3 * HH * DH * DH * 2);
    float*          hseq = (float*)        alloc((size_t)MROWS * DST * 4);
    unsigned short* yb   = (unsigned short*)alloc((size_t)MROWS * DST * 2);
    unsigned short* wop  = (unsigned short*)alloc((size_t)DST * DOUT * 2);
    (void)ws_size; (void)in_sizes; (void)n_in; (void)out_size;

    // 1. precision conversion / packing
    {
        size_t n = (size_t)MROWS * DIN;
        cvt_bf16_kernel<<<(unsigned)((n + 255) / 256), 256, 0, stream>>>(x, xb, n);
    }
    pack_b_kernel<<<(unsigned)(((size_t)DIN * PROJW + 255) / 256), 256, 0, stream>>>(
        w_in, wip, DIN, PROJW);
    pack_w_kernel<<<(3 * HH * DH * DH + 255) / 256, 256, 0, stream>>>(sw, wt);
    pack_b_kernel<<<(unsigned)(((size_t)DST * DOUT + 255) / 256), 256, 0, stream>>>(
        w_o, wop, DST, DOUT);

    // 2. input projection: proj = x @ w_in   (bf16 WMMA)
    gemm_bf16_kernel<<<dim3(PROJW / 128, MROWS / 128), 128, 0, stream>>>(
        xb, wip, proj, MROWS, PROJW, DIN);

    // 3. recurrent scan (4 waves per head, WMMA recurrence matmuls)
    scan_kernel<<<HH, 128, 0, stream>>>(proj, wt, hseq);

    // 4. gating + rmsnorm -> bf16 rows
    gate_norm_kernel<<<MROWS, 256, 0, stream>>>(hseq, proj, nw, yb);

    // 5. output projection: out = y @ w_out  (bf16 WMMA)
    gemm_bf16_kernel<<<dim3(DOUT / 128, MROWS / 128), 128, 0, stream>>>(
        yb, wop, out, MROWS, DOUT, DST);
}
